// Mamba_56831007261002
// MI455X (gfx1250) — matrix operand order repaired and checked by reading, not compiled, not run on hardware
//
#include <hip/hip_runtime.h>
#include <cstdint>

// ---------------------------------------------------------------------------
// MI455X (gfx1250) Mamba block: bf16 WMMA GEMMs + fp32 chunked complex scan.
// B=2 L=4096 D=1024 DM=2048 N=64 K=3  -> M = B*L = 8192 rows everywhere.
//
// ~330 GFLOP total (conv-as-3-GEMMs = 206 GFLOP dominates) vs ~0.5 GB HBM
// traffic -> compute bound; everything runs through v_wmma_f32_16x16x32_bf16
// (16384 FLOP/inst vs 2048 for the fp32 16x16x4 shape) with fp32 accumulate.
// GEMM pipeline: double-buffered LDS, ASYNCcnt DMA staging for the A tile.
// ---------------------------------------------------------------------------

typedef __bf16 bf16;
typedef __attribute__((ext_vector_type(16))) __bf16 v16bf;
typedef __attribute__((ext_vector_type(8)))  __bf16 v8bf;
typedef __attribute__((ext_vector_type(8)))  float  v8f;

#define BM 128
#define BN 128
#define BK 32
// LDS strides in bf16 elements: 56*2 = 112 bytes (16B aligned, 28-word row
// step -> 16 distinct banks across the 16 rows/cols a fragment touches).
#define LDS_A_STRIDE 56
#define LDS_B_STRIDE 56

// gfx1250 async global->LDS DMA (ASYNCcnt-tracked), if the toolchain has it.
#if defined(__HIP_DEVICE_COMPILE__) && \
    __has_builtin(__builtin_amdgcn_global_load_async_to_lds_b128) && \
    __has_builtin(__builtin_amdgcn_s_wait_asynccnt)
#define HAVE_ASYNC_LDS 1
#else
#define HAVE_ASYNC_LDS 0
#endif

#if HAVE_ASYNC_LDS
// Builtin prototype (from compiler diagnostic / upstream LLVM):
//   void __builtin_amdgcn_global_load_async_to_lds_b128(
//       v4i addrspace(1)* src, v4i addrspace(3)* dst, Imm int off, Imm int cpol)
typedef int v4i_vec __attribute__((vector_size(16)));
typedef __attribute__((address_space(1))) v4i_vec* gptr_v4i;
typedef __attribute__((address_space(3))) v4i_vec* lptr_v4i;

__device__ __forceinline__ void async_ld_b128(const void* g, void* l) {
  // Generic->LDS address: ISA 10.2 maps LDS aperture via addr[31:0] truncation.
  __builtin_amdgcn_global_load_async_to_lds_b128(
      (gptr_v4i)(uintptr_t)g,
      (lptr_v4i)(unsigned int)(uintptr_t)l,
      0, 0);
}
#endif

// ---------------- WMMA fragment loaders (ISA 7.12.2 layouts) ----------------

__device__ __forceinline__ v16bf make_v16(v8bf lo, v8bf hi) {
  v16bf f;
#pragma unroll
  for (int e = 0; e < 8; ++e) { f[e] = lo[e]; f[e + 8] = hi[e]; }
  return f;
}

// A fragment 16x32 (MxK): lane l holds row M=l&15;
// lanes 0-15: K {0..7, 16..23}; lanes 16-31: K {8..15, 24..31}.
__device__ __forceinline__ v16bf frag_a(const bf16* As, int mBase, int lane) {
  const int row = mBase + (lane & 15);
  const int kh  = (lane & 16) >> 1;   // 0 or 8
  const bf16* p = As + row * LDS_A_STRIDE;
  v8bf lo = *(const v8bf*)(p + kh);        // K = kh .. kh+7
  v8bf hi = *(const v8bf*)(p + 16 + kh);   // K = 16+kh .. 16+kh+7
  return make_v16(lo, hi);
}

// B fragment 32x16 (KxN), Bs stored transposed [col][k]:
// lane l holds col N=l&15; lanes 0-15: K {0..7, 16..23}; lanes 16-31: K {8..15, 24..31}.
__device__ __forceinline__ v16bf frag_b(const bf16* Bs, int nBase, int lane) {
  const int col = nBase + (lane & 15);
  const int kh  = (lane & 16) >> 1;   // 0 or 8
  const bf16* p = Bs + col * LDS_B_STRIDE + kh;
  v8bf lo = *(const v8bf*)(p);
  v8bf hi = *(const v8bf*)(p + 16);
  return make_v16(lo, hi);
}

// ---------------- Generic bf16 GEMM: C[M,N] = A[M,K] @ B[K,N] ---------------
// CONV: A rows are shift s-1 of x1 over L=4096 (zero-padded), B = Wconv[s],
//       flattened step loop over s in {0,1,2}.
// Epilogue: +bias, SiLU, multiply-by-aux (fp32 [M,N]), bf16/f32 out.

template <bool CONV, bool BF16OUT, bool SILU, bool MULAUX>
__global__ __launch_bounds__(256, 2)
void gemm_wmma(const bf16* __restrict__ A, const bf16* __restrict__ B,
               void* __restrict__ Cv, const float* __restrict__ bias,
               const float* __restrict__ aux, int M, int N, int K) {
  __shared__ bf16 As[2][BM * LDS_A_STRIDE];
  __shared__ bf16 Bs[2][BN * LDS_B_STRIDE];
  const int tid  = threadIdx.x;
  const int lane = tid & 31;
  const int wave = tid >> 5;
  const int wm   = wave & 3;   // 4 waves along M, 32 rows each
  const int wn   = wave >> 2;  // 2 waves along N, 64 cols each
  const int tileM = blockIdx.y * BM;
  const int tileN = blockIdx.x * BN;

  // Per-thread staging coordinates: 2 x 16B chunks each for A and B.
  const int ar0 = tid >> 2, ac0 = (tid & 3) << 3;   // A rows ar0, ar0+64
  const int kr0 = tid >> 4, cb0 = (tid & 15) << 3;  // B k-rows kr0, kr0+16

  const int stepsK = K / BK;
  const int total  = (CONV ? 3 : 1) * stepsK;

#if HAVE_ASYNC_LDS
  constexpr bool kAsync = !CONV;   // conv needs zero-fill on padded rows
#else
  constexpr bool kAsync = false;
#endif

  auto load_a = [&](int step, uint4* areg) {
    const int s  = CONV ? (step / stepsK) : 0;
    const int k0 = (CONV ? (step % stepsK) : step) * BK;
#pragma unroll
    for (int i = 0; i < 2; ++i) {
      const int gr = tileM + ar0 + 64 * i;
      uint4 v = make_uint4(0u, 0u, 0u, 0u);
      bool ok = true;
      int shift = 0;
      if (CONV) {
        const int ls = (gr & 4095) + s - 1;   // L = 4096
        ok = (ls >= 0 && ls < 4096);
        shift = s - 1;
      }
      if (ok)
        v = *(const uint4*)(A + (size_t)(gr + shift) * K + k0 + ac0);
      areg[i] = v;
    }
  };
  auto store_a = [&](int bufi, const uint4* areg) {
#pragma unroll
    for (int i = 0; i < 2; ++i)
      *(uint4*)&As[bufi][(ar0 + 64 * i) * LDS_A_STRIDE + ac0] = areg[i];
  };
  auto load_b = [&](int step, uint4* breg) {
    const int s  = CONV ? (step / stepsK) : 0;
    const int k0 = (CONV ? (step % stepsK) : step) * BK;
    const bf16* Bp = B + (CONV ? (size_t)s * K * N : (size_t)0);
#pragma unroll
    for (int i = 0; i < 2; ++i) {
      uint4 v = make_uint4(0u, 0u, 0u, 0u);
      if (tileN + cb0 < N)   // bounds for N=192 projection GEMM
        v = *(const uint4*)(Bp + (size_t)(k0 + kr0 + 16 * i) * N + tileN + cb0);
      breg[i] = v;
    }
  };
  auto store_b = [&](int bufi, const uint4* breg) {
#pragma unroll
    for (int i = 0; i < 2; ++i) {
      const bf16* tv = (const bf16*)&breg[i];
#pragma unroll
      for (int e = 0; e < 8; ++e)   // transpose into LDS [col][k]
        Bs[bufi][(cb0 + e) * LDS_B_STRIDE + (kr0 + 16 * i)] = tv[e];
    }
  };
#if HAVE_ASYNC_LDS
  auto issue_a_async = [&](int step, int bufi) {
    const int k0 = step * BK;
#pragma unroll
    for (int i = 0; i < 2; ++i)
      async_ld_b128(A + (size_t)(tileM + ar0 + 64 * i) * K + k0 + ac0,
                    &As[bufi][(ar0 + 64 * i) * LDS_A_STRIDE + ac0]);
  };
#endif

  v8f acc[2][4] = {};

  // Prologue: stage step 0 into buffer 0.
  {
    uint4 br[2];
    load_b(0, br);
    if constexpr (kAsync) {
#if HAVE_ASYNC_LDS
      issue_a_async(0, 0);
#endif
    } else {
      uint4 ar[2];
      load_a(0, ar);
      store_a(0, ar);
    }
    store_b(0, br);
  }

  int buf = 0;
  for (int step = 0; step < total; ++step) {
    const bool more = (step + 1) < total;
    if constexpr (kAsync) {
#if HAVE_ASYNC_LDS
      __builtin_amdgcn_s_wait_asynccnt(0);   // A DMA for `buf` complete
#endif
    }
    __syncthreads();                          // tile `buf` visible to all

    uint4 ar[2], br[2];
    (void)ar;
    if (more) {                               // start staging step+1 -> buf^1
      load_b(step + 1, br);
      if constexpr (kAsync) {
#if HAVE_ASYNC_LDS
        issue_a_async(step + 1, buf ^ 1);
#endif
      } else {
        load_a(step + 1, ar);
      }
    }

    // Compute on `buf`: 2x4 tiles of 16x16, K-slab of 32.
    v16bf af[2], bfr[4];
#pragma unroll
    for (int i = 0; i < 2; ++i) af[i] = frag_a(As[buf], wm * 32 + i * 16, lane);
#pragma unroll
    for (int j = 0; j < 4; ++j) bfr[j] = frag_b(Bs[buf], wn * 64 + j * 16, lane);
#pragma unroll
    for (int i = 0; i < 2; ++i)
#pragma unroll
      for (int j = 0; j < 4; ++j)
        acc[i][j] = __builtin_amdgcn_wmma_f32_16x16x32_bf16(
            false, af[i], false, bfr[j], (short)0, acc[i][j], false, false);

    if (more) {
      if constexpr (!kAsync) store_a(buf ^ 1, ar);
      store_b(buf ^ 1, br);
    }
    buf ^= 1;
  }

  // C/D layout: VGPR v -> M = v + 8*(lane>=16), N = lane&15.
  const int rOff = tileM + wm * 32 + ((lane & 16) >> 1);
  const int cOff = tileN + wn * 64 + (lane & 15);
#pragma unroll
  for (int i = 0; i < 2; ++i)
#pragma unroll
    for (int j = 0; j < 4; ++j)
#pragma unroll
      for (int v = 0; v < 8; ++v) {
        const int r = rOff + i * 16 + v;
        const int c = cOff + j * 16;
        if (c < N) {
          float val = acc[i][j][v];
          if (bias) val += bias[c];
          if (SILU) val = val / (1.f + __expf(-val));
          if (MULAUX) val *= aux[(size_t)r * N + c];
          if (BF16OUT) ((bf16*)Cv)[(size_t)r * N + c] = (bf16)val;
          else         ((float*)Cv)[(size_t)r * N + c] = val;
        }
      }
}

// -------------------------- small helper kernels ---------------------------

__global__ void k_f32_to_bf16(const float* __restrict__ in,
                              bf16* __restrict__ out, size_t n) {
  size_t i = (size_t)blockIdx.x * blockDim.x + threadIdx.x;
  const size_t stride = (size_t)gridDim.x * blockDim.x;
  for (; i < n; i += stride) out[i] = (bf16)in[i];
}

// [Wdelta | Wc_re | Wc_im] -> bf16 [2048][192]
__global__ void k_cat_proj(const float* __restrict__ Wd,
                           const float* __restrict__ Wcr,
                           const float* __restrict__ Wci,
                           bf16* __restrict__ out) {
  size_t i = (size_t)blockIdx.x * blockDim.x + threadIdx.x;
  const size_t stride = (size_t)gridDim.x * blockDim.x;
  const size_t n = (size_t)2048 * 192;
  for (; i < n; i += stride) {
    const int row = (int)(i / 192), j = (int)(i % 192);
    float v;
    if (j < 64)       v = Wd [row * 64 + j];
    else if (j < 128) v = Wcr[row * 64 + j - 64];
    else              v = Wci[row * 64 + j - 128];
    out[i] = (bf16)v;
  }
}

// [Wso_re ; -Wso_im] -> bf16 [128][2048]  (so hcat @ Wcat = Re(h @ Wso))
__global__ void k_cat_wso(const float* __restrict__ Wr,
                          const float* __restrict__ Wi,
                          bf16* __restrict__ out) {
  size_t i = (size_t)blockIdx.x * blockDim.x + threadIdx.x;
  const size_t stride = (size_t)gridDim.x * blockDim.x;
  const size_t n = (size_t)128 * 2048;
  for (; i < n; i += stride) {
    const int k = (int)(i / 2048), c = (int)(i % 2048);
    const float v = (k < 64) ? Wr[k * 2048 + c] : -Wi[(k - 64) * 2048 + c];
    out[i] = (bf16)v;
  }
}

// weights = exp(delta*(a+ib)); stage (w, xc) SoA for the scan.
__global__ void k_ssm_weights(const float* __restrict__ P,
                              const float* __restrict__ a,
                              const float* __restrict__ b,
                              float* __restrict__ Wre, float* __restrict__ Wim,
                              float* __restrict__ Hre, float* __restrict__ Him) {
  size_t i = (size_t)blockIdx.x * blockDim.x + threadIdx.x;
  const size_t stride = (size_t)gridDim.x * blockDim.x;
  const size_t n = (size_t)8192 * 64;
  for (; i < n; i += stride) {
    const int r = (int)(i >> 6), nn = (int)(i & 63);
    const float d  = P[r * 192 + nn];
    const float er = __expf(d * a[nn]);
    const float th = d * b[nn];
    Wre[i] = er * __cosf(th);
    Wim[i] = er * __sinf(th);
    Hre[i] = P[r * 192 + 64  + nn];
    Him[i] = P[r * 192 + 128 + nn];
  }
}

// --------- chunked complex scan: 128 sequences (b,n), 32 chunks of 128 -----
// combine: (w1,h1) . (w2,h2) = (w2*w1, w2*h1 + h2)

__global__ void k_scan_phase1(float* __restrict__ Wre, float* __restrict__ Wim,
                              float* __restrict__ Hre, float* __restrict__ Him) {
  const int t = threadIdx.x;            // 0..127 -> (b,n)
  const int b = t >> 6, n = t & 63;
  const int c = blockIdx.x;             // chunk 0..31
  float wr = 1.f, wi = 0.f, hr = 0.f, hi = 0.f;
  const size_t base = ((size_t)b * 4096 + (size_t)c * 128) * 64 + n;
  for (int l = 0; l < 128; ++l) {
    const size_t idx = base + (size_t)l * 64;
    const float w2r = Wre[idx], w2i = Wim[idx];
    const float x2r = Hre[idx], x2i = Him[idx];
    const float nhr = w2r * hr - w2i * hi + x2r;
    const float nhi = w2r * hi + w2i * hr + x2i;
    const float nwr = w2r * wr - w2i * wi;
    const float nwi = w2r * wi + w2i * wr;
    wr = nwr; wi = nwi; hr = nhr; hi = nhi;
    Wre[idx] = wr; Wim[idx] = wi; Hre[idx] = hr; Him[idx] = hi;
  }
}

__global__ void k_scan_phase2(const float* __restrict__ Wre,
                              const float* __restrict__ Wim,
                              const float* __restrict__ Hre,
                              const float* __restrict__ Him,
                              float* __restrict__ PrefR,
                              float* __restrict__ PrefI) {
  const int t = threadIdx.x;
  const int b = t >> 6, n = t & 63;
  float pr = 0.f, pi = 0.f;             // exclusive H prefix
  for (int c = 0; c < 32; ++c) {
    PrefR[c * 128 + t] = pr;
    PrefI[c * 128 + t] = pi;
    const size_t idx = ((size_t)b * 4096 + (size_t)c * 128 + 127) * 64 + n;
    const float wr = Wre[idx], wi = Wim[idx];
    const float hr = Hre[idx], hi = Him[idx];
    const float npr = wr * pr - wi * pi + hr;
    const float npi = wr * pi + wi * pr + hi;
    pr = npr; pi = npi;
  }
}

__global__ void k_scan_phase3(const float* __restrict__ Wre,
                              const float* __restrict__ Wim,
                              const float* __restrict__ Hre,
                              const float* __restrict__ Him,
                              const float* __restrict__ PrefR,
                              const float* __restrict__ PrefI,
                              bf16* __restrict__ hcat) {
  const int t = threadIdx.x;
  const int b = t >> 6, n = t & 63;
  const int c = blockIdx.x;
  const float pr = PrefR[c * 128 + t], pi = PrefI[c * 128 + t];
  const size_t base = ((size_t)b * 4096 + (size_t)c * 128) * 64 + n;
  for (int l = 0; l < 128; ++l) {
    const size_t idx = base + (size_t)l * 64;
    const float wr = Wre[idx], wi = Wim[idx];
    const float hr = Hre[idx], hi = Him[idx];
    const float gr = wr * pr - wi * pi + hr;
    const float gi = wr * pi + wi * pr + hi;
    const size_t row = (size_t)b * 4096 + (size_t)c * 128 + l;
    hcat[row * 128 + n]      = (bf16)gr;
    hcat[row * 128 + 64 + n] = (bf16)gi;
  }
}

// ---------------------------------------------------------------------------

extern "C" void kernel_launch(void* const* d_in, const int* in_sizes, int n_in,
                              void* d_out, int out_size, void* d_ws, size_t ws_size,
                              hipStream_t stream) {
  (void)in_sizes; (void)n_in; (void)out_size; (void)ws_size;

  const float* x      = (const float*)d_in[0];
  const float* W1     = (const float*)d_in[1];
  const float* b1     = (const float*)d_in[2];
  const float* W2     = (const float*)d_in[3];
  const float* b2     = (const float*)d_in[4];
  const float* Wconv  = (const float*)d_in[5];
  const float* bconv  = (const float*)d_in[6];
  const float* Wdelta = (const float*)d_in[7];
  const float* av     = (const float*)d_in[8];
  const float* bv     = (const float*)d_in[9];
  const float* Wc_re  = (const float*)d_in[10];
  const float* Wc_im  = (const float*)d_in[11];
  const float* Wso_re = (const float*)d_in[12];
  const float* Wso_im = (const float*)d_in[13];
  const float* Wout   = (const float*)d_in[14];
  const float* bout   = (const float*)d_in[15];

  char* ws = (char*)d_ws;
  size_t off = 0;
  auto alloc = [&](size_t bytes) -> void* {
    void* p = ws + off;
    off = (off + bytes + 255) & ~(size_t)255;
    return p;
  };

  const size_t M = 8192;
  bf16*  x_bf     = (bf16*) alloc(M * 1024 * 2);
  bf16*  W1_bf    = (bf16*) alloc((size_t)1024 * 2048 * 2);
  bf16*  W2_bf    = (bf16*) alloc((size_t)1024 * 2048 * 2);
  bf16*  Wconv_bf = (bf16*) alloc((size_t)3 * 2048 * 2048 * 2);
  bf16*  Wproj_bf = (bf16*) alloc((size_t)2048 * 192 * 2);
  bf16*  Wso_bf   = (bf16*) alloc((size_t)128 * 2048 * 2);
  bf16*  Wout_bf  = (bf16*) alloc((size_t)2048 * 1024 * 2);
  bf16*  x1_bf    = (bf16*) alloc(M * 2048 * 2);
  float* x2s      = (float*)alloc(M * 2048 * 4);
  bf16*  x1c_bf   = (bf16*) alloc(M * 2048 * 2);
  float* P        = (float*)alloc(M * 192 * 4);
  float* Wre      = (float*)alloc(M * 64 * 4);
  float* Wim      = (float*)alloc(M * 64 * 4);
  float* Hre      = (float*)alloc(M * 64 * 4);
  float* Him      = (float*)alloc(M * 64 * 4);
  float* PrefR    = (float*)alloc((size_t)32 * 128 * 4);
  float* PrefI    = (float*)alloc((size_t)32 * 128 * 4);
  bf16*  hcat     = (bf16*) alloc(M * 128 * 2);
  bf16*  g_bf     = (bf16*) alloc(M * 2048 * 2);

  // ---- precision staging (fp32 -> bf16) ----
  k_f32_to_bf16<<<2048, 256, 0, stream>>>(x,     x_bf,     M * 1024);
  k_f32_to_bf16<<<1024, 256, 0, stream>>>(W1,    W1_bf,    (size_t)1024 * 2048);
  k_f32_to_bf16<<<1024, 256, 0, stream>>>(W2,    W2_bf,    (size_t)1024 * 2048);
  k_f32_to_bf16<<<2048, 256, 0, stream>>>(Wconv, Wconv_bf, (size_t)3 * 2048 * 2048);
  k_f32_to_bf16<<<1024, 256, 0, stream>>>(Wout,  Wout_bf,  (size_t)2048 * 1024);
  k_cat_proj<<<512, 256, 0, stream>>>(Wdelta, Wc_re, Wc_im, Wproj_bf);
  k_cat_wso<<<512, 256, 0, stream>>>(Wso_re, Wso_im, Wso_bf);

  // ---- x1 = x@W1 + b1 (bf16 out), x2s = silu(x@W2 + b2) (fp32) ----
  dim3 gIn(2048 / BN, M / BM);
  gemm_wmma<false, true,  false, false><<<gIn, 256, 0, stream>>>(
      x_bf, W1_bf, x1_bf, b1, nullptr, (int)M, 2048, 1024);
  gemm_wmma<false, false, true,  false><<<gIn, 256, 0, stream>>>(
      x_bf, W2_bf, x2s, b2, nullptr, (int)M, 2048, 1024);

  // ---- x1c = silu(conv1d(x1) + bconv) as 3 shifted GEMM passes ----
  gemm_wmma<true, true, true, false><<<gIn, 256, 0, stream>>>(
      x1_bf, Wconv_bf, x1c_bf, bconv, nullptr, (int)M, 2048, 2048);

  // ---- P = x1c @ [Wdelta | Wc_re | Wc_im]  ([8192,192], fp32) ----
  dim3 gP((192 + BN - 1) / BN, M / BM);
  gemm_wmma<false, false, false, false><<<gP, 256, 0, stream>>>(
      x1c_bf, Wproj_bf, P, nullptr, nullptr, (int)M, 192, 2048);

  // ---- selective-scan: weights + chunked complex prefix scan ----
  k_ssm_weights<<<2048, 256, 0, stream>>>(P, av, bv, Wre, Wim, Hre, Him);
  k_scan_phase1<<<32, 128, 0, stream>>>(Wre, Wim, Hre, Him);
  k_scan_phase2<<<1, 128, 0, stream>>>(Wre, Wim, Hre, Him, PrefR, PrefI);
  k_scan_phase3<<<32, 128, 0, stream>>>(Wre, Wim, Hre, Him, PrefR, PrefI, hcat);

  // ---- g = Re(h @ Wso) * silu(x2)  (fused aux multiply, bf16 out) ----
  gemm_wmma<false, true, false, true><<<gIn, 256, 0, stream>>>(
      hcat, Wso_bf, g_bf, nullptr, x2s, (int)M, 2048, 128);

  // ---- y = g @ Wout + bout (fp32 to d_out) ----
  dim3 gOut(1024 / BN, M / BM);
  gemm_wmma<false, false, false, false><<<gOut, 256, 0, stream>>>(
      g_bf, Wout_bf, d_out, bout, nullptr, (int)M, 1024, 2048);
}
